// TwoStageMMoEModel_40931038331294
// MI455X (gfx1250) — matrix operand order repaired and checked by reading, not compiled, not run on hardware
//
#include <hip/hip_runtime.h>

typedef _Float16 half_t;
typedef __attribute__((ext_vector_type(16))) _Float16 v16h;
typedef __attribute__((ext_vector_type(8)))  _Float16 v8h;
typedef __attribute__((ext_vector_type(8)))  float    v8f;
typedef __attribute__((ext_vector_type(2)))  float    v2f;

#define BN_SCALE 0.9999950000374997f  // 1/sqrt(1+1e-5)

// ---- input index map (setup_inputs dict insertion order, params flattened depth-first) ----
enum {
  I_UE = 0, I_IE, I_DE, I_WE, I_YE,                 // 0..4 embedding tables
  I_UB, I_IB, I_DB, I_WB, I_YB, I_UTB, I_GB,        // 5..11 biases
  I_T1W, I_T1B, I_TG1, I_TB1, I_T2W, I_T2B, I_TG2, I_TB2, I_T3W, I_T3B, I_T4W, I_T4B, // 12..23
  I_C1W, I_C1B, I_CG1, I_CB1, I_C2W, I_C2B, I_C3W, I_C3B,                             // 24..31
  I_RW, I_RB,                                                                          // 32..33
  I_EW1, I_EB1, I_EBG, I_EBB, I_EW2, I_EB2, I_EW3, I_EB3,                              // 34..41
  I_G1W, I_G1B, I_GG1, I_GB1, I_G2W, I_G2B, I_G3W, I_G3B,                              // 42..49
  I_F1W, I_F1B, I_FG1, I_FB1, I_F2W, I_F2B, I_F3W, I_F3B,                              // 50..57
  I_FW,                                                                                // 58
  I_UIDX, I_IIDX, I_DIDX, I_WIDX, I_YIDX                                               // 59..63
};

// -------------------------------------------------------------------------
// Weight packer: fp32 [K,N] row-major -> f16 WMMA B-fragments.
// Fragment (nt,kt) = 512 halves; lane L holds N = nt*16 + L%16,
// K = kt*32 + (j/8)*16 + (L/16)*8 + j%8, j=0..15, contiguous at lane*32 bytes.
// -------------------------------------------------------------------------
__global__ void pack_w_kernel(const float* __restrict__ src, half_t* __restrict__ dst,
                              int K, int N, int kT, int nT) {
  int idx = blockIdx.x * 256 + threadIdx.x;
  int total = kT * nT * 512;
  if (idx >= total) return;
  int f = idx >> 9;
  int w = idx & 511;
  int l = w >> 4;
  int j = w & 15;
  int nt = f / kT, kt = f - nt * kT;
  int n = nt * 16 + (l & 15);
  int k = kt * 32 + ((j >> 3) << 4) + ((l >> 4) << 3) + (j & 7);
  float v = (k < K && n < N) ? src[(size_t)k * N + n] : 0.0f;
  dst[idx] = (half_t)v;
}

struct MP {
  const float *ue, *ie, *de, *we, *ye;
  const float *ub, *ib, *db, *wb, *yb, *utb, *gb;
  const float *t1b, *tg1, *tb1, *t2b, *tg2, *tb2, *t3b, *t4b;
  const float *c1b, *cg1, *cb1, *c2b, *c3b;
  const float *rw, *rb;
  const float *ew1, *eb1, *ebg, *ebb, *eb2, *eb3;
  const float *g1b, *gg1, *gb1, *g2b, *g3b;
  const float *f1b, *fg1, *fb1, *f2b, *f3b;
  const float *fw;
  const int *ui, *ii, *di, *wi, *yi;
  const half_t *pT1, *pT2, *pT3, *pT4, *pC1, *pC2, *pC3, *pG1, *pG2, *pG3, *pE2, *pE3, *pF1, *pF2, *pF3;
};

// -------------------------------------------------------------------------
// One GEMM layer for a 16-row batch tile: [16,K] (LDS f16) x [K,N] (packed
// global f16) -> f32, epilogue mode 0=linear 1=relu 2=relu+BN, result to
// LDS f16 (actOut) and/or LDS f32 (f32Out, only cols < realN).
// -------------------------------------------------------------------------
__device__ __forceinline__ void mlp_layer(int lane,
    const half_t* actIn, int inStride,
    half_t* actOut, int outStride,
    float* f32Out, int f32Stride,
    const half_t* __restrict__ Wp, int kT, int nT,
    const float* __restrict__ bias,
    const float* __restrict__ bn_g, const float* __restrict__ bn_b,
    int mode, int realN)
{
  const int nl   = lane & 15;
  const int hi   = lane >> 4;
  const int row0 = hi * 8;
  for (int nt = 0; nt < nT; ++nt) {
    const int n = nt * 16 + nl;
    float bv = (bias && n < realN) ? bias[n] : 0.0f;
    v8f acc;
#pragma unroll
    for (int r = 0; r < 8; ++r) acc[r] = bv;
    for (int kt = 0; kt < kT; ++kt) {
      // A fragment (16-bit A layout: lane row = L%16, K split 8 + 8)
      const half_t* ap = actIn + nl * inStride + kt * 32 + hi * 8;
      union { v16h v; v8h h[2]; } A;
      A.h[0] = *(const v8h*)(ap);
      A.h[1] = *(const v8h*)(ap + 16);
      // B fragment from packed weights: 32 contiguous bytes per lane
      v16h Bv = *(const v16h*)(Wp + ((size_t)(nt * kT + kt) << 9) + (lane << 4));
      acc = __builtin_amdgcn_wmma_f32_16x16x32_f16(false, A.v, false, Bv,
                                                   (short)0, acc, false, false);
    }
    float g = 1.0f, bb = 0.0f;
    if (mode == 2 && n < realN) { g = bn_g[n] * BN_SCALE; bb = bn_b[n]; }
#pragma unroll
    for (int r = 0; r < 8; ++r) {
      float x = acc[r];
      if (mode >= 1) x = fmaxf(x, 0.0f);
      if (mode == 2) x = g * x + bb;
      acc[r] = x;
    }
    if (actOut) {
#pragma unroll
      for (int r = 0; r < 8; ++r) actOut[(row0 + r) * outStride + n] = (half_t)acc[r];
    }
    if (f32Out && n < realN) {
#pragma unroll
      for (int r = 0; r < 8; ++r) f32Out[(row0 + r) * f32Stride + n] = acc[r];
    }
  }
}

// -------------------------------------------------------------------------
// Main fused kernel: one wave (32 threads) per 16 batch rows.
// -------------------------------------------------------------------------
__global__ __launch_bounds__(32) void mmoe_kernel(MP P, float* __restrict__ out, int B)
{
  const int lane = threadIdx.x;
  const int base = blockIdx.x * 16;

  __shared__ alignas(16) half_t actA[16 * 288];  // [ue|ie|de|we|ye|pad]
  __shared__ alignas(16) half_t actG[16 * 192];  // [ue|tf|tp|cp|pad]
  __shared__ alignas(16) half_t H1[16 * 256];
  __shared__ alignas(16) half_t H2[16 * 128];
  __shared__ float eoutS[4 * 16 * 16];
  __shared__ float einS[16 * 4];
  __shared__ float gatesS[16 * 4];
  __shared__ float tpS[16], tiS[16], cpS[16], moS[16];
  __shared__ int uIdx[16], iIdx[16], dIdx[16], wIdx[16], yIdx[16];

  if (lane < 16) {
    int b = base + lane;
    uIdx[lane] = P.ui[b]; iIdx[lane] = P.ii[b]; dIdx[lane] = P.di[b];
    wIdx[lane] = P.wi[b]; yIdx[lane] = P.yi[b];
  }
  __syncthreads();

  // gather embeddings -> actA [16][288] f16 (cols 260..287 zero)
  for (int i = lane; i < 16 * 288; i += 32) {
    int r = i / 288, c = i - r * 288;
    float v = 0.0f;
    if (c < 100)      v = P.ue[(size_t)uIdx[r] * 100 + c];
    else if (c < 200) v = P.ie[(size_t)iIdx[r] * 100 + (c - 100)];
    else if (c < 220) v = P.de[dIdx[r] * 20 + (c - 200)];
    else if (c < 240) v = P.we[wIdx[r] * 20 + (c - 220)];
    else if (c < 260) v = P.ye[yIdx[r] * 20 + (c - 240)];
    actA[i] = (half_t)v;
  }
  __syncthreads();

  // actG: cols 0..99 = ue, 100..159 = [de|we|ye], 160/161 = tp/cp (later), rest 0
  for (int i = lane; i < 16 * 192; i += 32) {
    int r = i / 192, c = i - r * 192;
    half_t v = (half_t)0.0f;
    if (c < 100)      v = actA[r * 288 + c];
    else if (c < 160) v = actA[r * 288 + 100 + c];
    actG[i] = v;
  }

  // scalar CF pieces: base dot, residual, all bias gathers
  if (lane < 16) {
    int r = lane;
    float s = 0.0f, res = 0.0f;
    for (int k = 0; k < 100; ++k) {
      float u = (float)actA[r * 288 + k];
      s   += u * (float)actA[r * 288 + 100 + k];
      res += u * P.rw[k];
    }
    cpS[r] = s + res + P.rb[0]
           + P.ub[uIdx[r]] + P.ib[iIdx[r]] + P.db[dIdx[r]]
           + P.wb[wIdx[r]] + P.yb[yIdx[r]]
           + P.utb[uIdx[r] * 3 + dIdx[r]] + P.gb[0];
  }
  __syncthreads();

  // ---- temporal MLP: 260 -> 256 -> 128 -> 64 -> 1 ----
  mlp_layer(lane, actA, 288, H1, 256, nullptr, 0, P.pT1, 9, 16, P.t1b, P.tg1, P.tb1, 2, 256);
  __syncthreads();
  mlp_layer(lane, H1, 256, H2, 128, nullptr, 0, P.pT2, 8, 8, P.t2b, P.tg2, P.tb2, 2, 128);
  __syncthreads();
  mlp_layer(lane, H2, 128, H1, 64, nullptr, 0, P.pT3, 4, 4, P.t3b, nullptr, nullptr, 1, 64);
  __syncthreads();
  mlp_layer(lane, H1, 64, nullptr, 0, tpS, 1, P.pT4, 2, 1, P.t4b, nullptr, nullptr, 0, 1);
  __syncthreads();

  // ---- cf_fusion MLP: 160 -> 128 -> 64 -> 1 ----
  mlp_layer(lane, actG, 192, H2, 128, nullptr, 0, P.pC1, 5, 8, P.c1b, P.cg1, P.cb1, 2, 128);
  __syncthreads();
  mlp_layer(lane, H2, 128, H1, 64, nullptr, 0, P.pC2, 4, 4, P.c2b, nullptr, nullptr, 1, 64);
  __syncthreads();
  mlp_layer(lane, H1, 64, nullptr, 0, tiS, 1, P.pC3, 2, 1, P.c3b, nullptr, nullptr, 0, 1);
  __syncthreads();

  if (lane < 16) {
    float tp = tpS[lane];
    float cp = cpS[lane] + tiS[lane];
    cpS[lane] = cp;
    actG[lane * 192 + 160] = (half_t)tp;
    actG[lane * 192 + 161] = (half_t)cp;
    einS[lane * 4 + 0] = tp;
    einS[lane * 4 + 1] = cp;
    einS[lane * 4 + 2] = tp - cp;
    einS[lane * 4 + 3] = tp * cp;
  }
  __syncthreads();

  // ---- gate MLP: 162 -> 128 -> 64 -> 4, softmax ----
  mlp_layer(lane, actG, 192, H2, 128, nullptr, 0, P.pG1, 6, 8, P.g1b, P.gg1, P.gb1, 2, 128);
  __syncthreads();
  mlp_layer(lane, H2, 128, H1, 64, nullptr, 0, P.pG2, 4, 4, P.g2b, nullptr, nullptr, 1, 64);
  __syncthreads();
  mlp_layer(lane, H1, 64, nullptr, 0, gatesS, 4, P.pG3, 2, 1, P.g3b, nullptr, nullptr, 0, 4);
  __syncthreads();

  if (lane < 16) {
    float a = gatesS[lane*4+0], b = gatesS[lane*4+1], c = gatesS[lane*4+2], d = gatesS[lane*4+3];
    float mx = fmaxf(fmaxf(a, b), fmaxf(c, d));
    float ea = __expf(a - mx), eb = __expf(b - mx), ec = __expf(c - mx), ed = __expf(d - mx);
    float inv = 1.0f / (ea + eb + ec + ed);
    gatesS[lane*4+0] = ea * inv; gatesS[lane*4+1] = eb * inv;
    gatesS[lane*4+2] = ec * inv; gatesS[lane*4+3] = ed * inv;
  }
  __syncthreads();

  // ---- experts: ein[16,4] -> 64 (f32 K=4 WMMA, relu+BN) -> 32 -> 16 ----
  {
    const int nl = lane & 15, hi = lane >> 4, row0 = hi * 8;
    const int k0 = hi * 2;  // f32 16x16x4 A layout: lanes<16 hold K0,K1; lanes>=16 hold K2,K3
    v2f a;
    a[0] = einS[nl * 4 + k0];
    a[1] = einS[nl * 4 + k0 + 1];
    for (int e = 0; e < 4; ++e) {
      for (int nt = 0; nt < 4; ++nt) {
        int n = nt * 16 + nl;
        v2f bf;
        bf[0] = P.ew1[((size_t)e * 4 + k0) * 64 + n];
        bf[1] = P.ew1[((size_t)e * 4 + k0 + 1) * 64 + n];
        float bv = P.eb1[e * 64 + n];
        v8f acc;
#pragma unroll
        for (int r = 0; r < 8; ++r) acc[r] = bv;
        acc = __builtin_amdgcn_wmma_f32_16x16x4_f32(false, a, false, bf,
                                                    (short)0, acc, false, false);
        float g = P.ebg[e * 64 + n] * BN_SCALE, bb = P.ebb[e * 64 + n];
#pragma unroll
        for (int r = 0; r < 8; ++r) {
          float x = fmaxf(acc[r], 0.0f);
          H2[(row0 + r) * 64 + n] = (half_t)(g * x + bb);
        }
      }
      __syncthreads();
      mlp_layer(lane, H2, 64, H1, 32, nullptr, 0, P.pE2 + e * 2048, 2, 2,
                P.eb2 + e * 32, nullptr, nullptr, 1, 32);
      __syncthreads();
      mlp_layer(lane, H1, 32, nullptr, 0, eoutS + e * 256, 16, P.pE3 + e * 512, 1, 1,
                P.eb3 + e * 16, nullptr, nullptr, 0, 16);
      __syncthreads();
    }
  }

  // mixed = gates . eout  -> H2 [16][32] f16 (cols 16..31 zero)
  for (int i = lane; i < 16 * 32; i += 32) {
    int r = i >> 5, c = i & 31;
    float v = 0.0f;
    if (c < 16) {
#pragma unroll
      for (int e = 0; e < 4; ++e) v += gatesS[r * 4 + e] * eoutS[e * 256 + r * 16 + c];
    }
    H2[i] = (half_t)v;
  }
  __syncthreads();

  // ---- final MLP: 16 -> 64 -> 32 -> 1 ----
  mlp_layer(lane, H2, 32, H1, 64, nullptr, 0, P.pF1, 1, 4, P.f1b, P.fg1, P.fb1, 2, 64);
  __syncthreads();
  mlp_layer(lane, H1, 64, H2, 32, nullptr, 0, P.pF2, 2, 2, P.f2b, nullptr, nullptr, 1, 32);
  __syncthreads();
  mlp_layer(lane, H2, 32, nullptr, 0, moS, 1, P.pF3, 1, 1, P.f3b, nullptr, nullptr, 0, 1);
  __syncthreads();

  if (lane < 16 && base + lane < B) {
    float w0 = P.fw[0], w1 = P.fw[1];
    float mx = fmaxf(w0, w1);
    float e0 = __expf(w0 - mx), e1 = __expf(w1 - mx);
    float inv = 1.0f / (e0 + e1);
    out[base + lane] = (e0 * inv) * tpS[lane] + (e1 * inv) * cpS[lane] + moS[lane];
  }
}

// -------------------------------------------------------------------------
extern "C" void kernel_launch(void* const* d_in, const int* in_sizes, int n_in,
                              void* d_out, int out_size, void* d_ws, size_t ws_size,
                              hipStream_t stream) {
  auto f  = [&](int i) { return (const float*)d_in[i]; };
  auto ip = [&](int i) { return (const int*)d_in[i]; };

  half_t* ws = (half_t*)d_ws;
  size_t off = 0;
  auto region = [&](int kT, int nT, int count) {
    half_t* p = ws + off;
    off += (size_t)kT * nT * 512 * count;
    return p;
  };
  half_t* pT1 = region(9, 16, 1);
  half_t* pT2 = region(8, 8, 1);
  half_t* pT3 = region(4, 4, 1);
  half_t* pT4 = region(2, 1, 1);
  half_t* pC1 = region(5, 8, 1);
  half_t* pC2 = region(4, 4, 1);
  half_t* pC3 = region(2, 1, 1);
  half_t* pG1 = region(6, 8, 1);
  half_t* pG2 = region(4, 4, 1);
  half_t* pG3 = region(2, 1, 1);
  half_t* pE2 = region(2, 2, 4);
  half_t* pE3 = region(1, 1, 4);
  half_t* pF1 = region(1, 4, 1);
  half_t* pF2 = region(2, 2, 1);
  half_t* pF3 = region(1, 1, 1);

  auto pack = [&](const float* src, half_t* dst, int K, int N, int kT, int nT) {
    int total = kT * nT * 512;
    hipLaunchKernelGGL(pack_w_kernel, dim3((total + 255) / 256), dim3(256), 0, stream,
                       src, dst, K, N, kT, nT);
  };
  pack(f(I_T1W), pT1, 260, 256, 9, 16);
  pack(f(I_T2W), pT2, 256, 128, 8, 8);
  pack(f(I_T3W), pT3, 128, 64, 4, 4);
  pack(f(I_T4W), pT4, 64, 1, 2, 1);
  pack(f(I_C1W), pC1, 160, 128, 5, 8);
  pack(f(I_C2W), pC2, 128, 64, 4, 4);
  pack(f(I_C3W), pC3, 64, 1, 2, 1);
  pack(f(I_G1W), pG1, 162, 128, 6, 8);
  pack(f(I_G2W), pG2, 128, 64, 4, 4);
  pack(f(I_G3W), pG3, 64, 4, 2, 1);
  for (int e = 0; e < 4; ++e) {
    pack(f(I_EW2) + (size_t)e * 64 * 32, pE2 + (size_t)e * 2048, 64, 32, 2, 2);
    pack(f(I_EW3) + (size_t)e * 32 * 16, pE3 + (size_t)e * 512, 32, 16, 1, 1);
  }
  pack(f(I_F1W), pF1, 16, 64, 1, 4);
  pack(f(I_F2W), pF2, 64, 32, 2, 2);
  pack(f(I_F3W), pF3, 32, 1, 1, 1);

  MP P;
  P.ue = f(I_UE); P.ie = f(I_IE); P.de = f(I_DE); P.we = f(I_WE); P.ye = f(I_YE);
  P.ub = f(I_UB); P.ib = f(I_IB); P.db = f(I_DB); P.wb = f(I_WB); P.yb = f(I_YB);
  P.utb = f(I_UTB); P.gb = f(I_GB);
  P.t1b = f(I_T1B); P.tg1 = f(I_TG1); P.tb1 = f(I_TB1);
  P.t2b = f(I_T2B); P.tg2 = f(I_TG2); P.tb2 = f(I_TB2);
  P.t3b = f(I_T3B); P.t4b = f(I_T4B);
  P.c1b = f(I_C1B); P.cg1 = f(I_CG1); P.cb1 = f(I_CB1);
  P.c2b = f(I_C2B); P.c3b = f(I_C3B);
  P.rw = f(I_RW); P.rb = f(I_RB);
  P.ew1 = f(I_EW1); P.eb1 = f(I_EB1); P.ebg = f(I_EBG); P.ebb = f(I_EBB);
  P.eb2 = f(I_EB2); P.eb3 = f(I_EB3);
  P.g1b = f(I_G1B); P.gg1 = f(I_GG1); P.gb1 = f(I_GB1);
  P.g2b = f(I_G2B); P.g3b = f(I_G3B);
  P.f1b = f(I_F1B); P.fg1 = f(I_FG1); P.fb1 = f(I_FB1);
  P.f2b = f(I_F2B); P.f3b = f(I_F3B);
  P.fw = f(I_FW);
  P.ui = ip(I_UIDX); P.ii = ip(I_IIDX); P.di = ip(I_DIDX); P.wi = ip(I_WIDX); P.yi = ip(I_YIDX);
  P.pT1 = pT1; P.pT2 = pT2; P.pT3 = pT3; P.pT4 = pT4;
  P.pC1 = pC1; P.pC2 = pC2; P.pC3 = pC3;
  P.pG1 = pG1; P.pG2 = pG2; P.pG3 = pG3;
  P.pE2 = pE2; P.pE3 = pE3;
  P.pF1 = pF1; P.pF2 = pF2; P.pF3 = pF3;

  int B = in_sizes[I_UIDX];
  int tiles = (B + 15) / 16;
  hipLaunchKernelGGL(mmoe_kernel, dim3(tiles), dim3(32), 0, stream, P, (float*)d_out, B);
}
